// RIB_58110907515253
// MI455X (gfx1250) — compile-verified, matrix operand order checked by reading, not run on hardware
//
#include <hip/hip_runtime.h>

// ---------------------------------------------------------------------------
// Types / helpers
// ---------------------------------------------------------------------------
typedef __attribute__((ext_vector_type(16))) __bf16 v16bf;
typedef __attribute__((ext_vector_type(8)))  float  v8f;
typedef int vsi4 __attribute__((vector_size(16)));

union FragBF { v16bf v; unsigned short u[16]; uint4 q[2]; };

__device__ __forceinline__ unsigned short f2bf_u(float f) {
  union { float f; unsigned u; } v; v.f = f;
  unsigned u = v.u;
  unsigned r = (u + 0x7fffu + ((u >> 16) & 1u)) >> 16;  // RNE
  return (unsigned short)r;
}

#define LEAKY(y) ((y) >= 0.f ? (y) : 0.01f * (y))
#define BN_EPS 1e-5f

static const int NB   = 2;
static const int HH   = 50;
static const int WW   = 100;
static const int NPix = 5000;   // H*W
static const int CMAX = 1024;   // cluster cap
static const int MMAX = 1024;   // representative cap
static const int KTOP = 8;

// -- gfx1250 async global->LDS (ASYNCcnt) with safe fallback ----------------
__device__ __forceinline__ void async_copy_b128(unsigned short* ldst,
                                                const unsigned short* gsrc) {
#if __has_builtin(__builtin_amdgcn_global_load_async_to_lds_b128)
  __builtin_amdgcn_global_load_async_to_lds_b128(
      (__attribute__((address_space(1))) vsi4*)gsrc,
      (__attribute__((address_space(3))) vsi4*)ldst, 0, 0);
#else
  *(uint4*)ldst = *(const uint4*)gsrc;
#endif
}
__device__ __forceinline__ void async_wait_all() {
#if __has_builtin(__builtin_amdgcn_s_wait_asynccnt)
  __builtin_amdgcn_s_wait_asynccnt(0);
#endif
}

// ---------------------------------------------------------------------------
// Trivial utility kernels
// ---------------------------------------------------------------------------
__global__ void k_fill_f32(float* p, float v, long n) {
  long i = (long)blockIdx.x * blockDim.x + threadIdx.x;
  if (i < n) p[i] = v;
}
__global__ void k_fill_i32(int* p, int v, long n) {
  long i = (long)blockIdx.x * blockDim.x + threadIdx.x;
  if (i < n) p[i] = v;
}
__global__ void k_f32_to_bf16(unsigned short* dst, const float* src, long n) {
  long i = (long)blockIdx.x * blockDim.x + threadIdx.x;
  if (i < n) dst[i] = f2bf_u(src[i]);
}
__global__ void k_add_f32(float* dst, const float* a, const float* b, long n) {
  long i = (long)blockIdx.x * blockDim.x + threadIdx.x;
  if (i < n) dst[i] = a[i] + b[i];
}
// dst[c*R + r] = bf16(src[r*C + c])   (src row-major [R][C])
__global__ void k_transpose_bf(unsigned short* dst, const float* src, int R, int C) {
  long idx = (long)blockIdx.x * blockDim.x + threadIdx.x;
  if (idx >= (long)R * C) return;
  int r = (int)(idx / C), c = (int)(idx % C);
  dst[(size_t)c * R + r] = f2bf_u(src[idx]);
}

// ---------------------------------------------------------------------------
// Weight packing
// ---------------------------------------------------------------------------
// w [Cout][Cin][3][3] f32  ->  wp [(tap*Cin+ci)][Cout] bf16  (tap = kh*3+kw)
__global__ void k_pack_conv_w(unsigned short* wp, const float* w, int Cout, int Cin) {
  long idx = (long)blockIdx.x * blockDim.x + threadIdx.x;
  long total = (long)Cout * Cin * 9;
  if (idx >= total) return;
  int co  = (int)(idx / ((long)Cin * 9));
  int rem = (int)(idx % ((long)Cin * 9));
  int ci  = rem / 9;
  int tap = rem % 9;
  wp[((size_t)tap * Cin + ci) * Cout + co] = f2bf_u(w[idx]);
}
// seq conv: only kw==1 column survives (W'=1, SAME). wps[(dt*512+ci)][co] f32
__global__ void k_pack_seq_w(float* wps, const float* w) {
  long idx = (long)blockIdx.x * blockDim.x + threadIdx.x;
  if (idx >= (long)512 * 512 * 3) return;
  int co  = (int)(idx / (512 * 3));
  int rem = (int)(idx % (512 * 3));
  int ci  = rem / 3;
  int dt  = rem % 3;
  wps[((size_t)dt * 512 + ci) * 512 + co] = w[(((size_t)co * 512 + ci) * 3 + dt) * 3 + 1];
}

// ---------------------------------------------------------------------------
// Implicit-GEMM 3x3 conv + BN + leaky.
// Block = 8 waves stacked along pixel tiles sharing one 32-channel tile.
// Weight tile (32x32 bf16 = 2KB) staged in LDS via async global->LDS.
// Each wave computes TWO 16x16 output tiles per A fragment (2 WMMA / k-step).
// x bf16 [nb][Cin][N], wp bf16 [9*Cin][Cout], out f32 [nb][Cout][N]
// Cout must be a multiple of 32 (it is 512 everywhere here).
// ---------------------------------------------------------------------------
__global__ __launch_bounds__(256) void k_conv_bn(
    float* __restrict__ out, const unsigned short* __restrict__ x,
    const unsigned short* __restrict__ wp,
    const float* __restrict__ g, const float* __restrict__ bb,
    const float* __restrict__ mm, const float* __restrict__ vv,
    const unsigned short* __restrict__ zpage,
    int nbatch, int Cin, int Cout, int H, int W) {
  const int N   = H * W;
  const int TM  = (N + 15) >> 4;
  const int TMB = (TM + 7) >> 3;
  const int TN  = Cout >> 5;            // 32-channel tiles
  __shared__ unsigned short ldsB[32 * 32];

  int lane = threadIdx.x & 31;
  int wloc = threadIdx.x >> 5;          // 0..7
  long blk = blockIdx.x;
  int b   = (int)(blk / ((long)TMB * TN));
  int r0  = (int)(blk % ((long)TMB * TN));
  int tmb = r0 / TN, tn = r0 % TN;
  int tm  = tmb * 8 + wloc;

  int mrow = lane & 15, half = lane >> 4;
  int pix = tm * 16 + mrow;
  bool pvalid = (tm < TM) && (pix < N);
  int ph = pvalid ? pix / W : 0;
  int pw = pvalid ? pix % W : 0;
  int c0   = tn * 32;
  int cch0 = c0 + mrow;
  int cch1 = c0 + 16 + mrow;
  float scale0 = g[cch0] * rsqrtf(vv[cch0] + BN_EPS);
  float bias0  = bb[cch0] - mm[cch0] * scale0;
  float scale1 = g[cch1] * rsqrtf(vv[cch1] + BN_EPS);
  float bias1  = bb[cch1] - mm[cch1] * scale1;
  const unsigned short* xb = x + (size_t)b * Cin * N;

  // staging role: threads 0..127 each move one 16-byte segment per k-step
  int st_t  = threadIdx.x;
  int st_r  = st_t >> 2;                // 0..31 row (only for t<128)
  int st_sg = st_t & 3;                 // 0..3 segment (8 cols each)

  v8f acc0 = {0.f, 0.f, 0.f, 0.f, 0.f, 0.f, 0.f, 0.f};
  v8f acc1 = {0.f, 0.f, 0.f, 0.f, 0.f, 0.f, 0.f, 0.f};
  for (int tap = 0; tap < 9; ++tap) {
    int kh = tap / 3, kw = tap % 3;
    int hs = ph + kh - 1, ws2 = pw + kw - 1;
    bool inb = pvalid && hs >= 0 && hs < H && ws2 >= 0 && ws2 < W;
    const unsigned short* abase = inb ? (xb + hs * W + ws2) : zpage;
    size_t astr = inb ? (size_t)N : 0;
    for (int cb = 0; cb < Cin; cb += 32) {
      int k0 = tap * Cin + cb;
      __syncthreads();                  // previous LDS reads done
      if (st_t < 128) {
        async_copy_b128(&ldsB[st_r * 32 + st_sg * 8],
                        &wp[(size_t)(k0 + st_r) * Cout + c0 + st_sg * 8]);
      }
      async_wait_all();
      __syncthreads();                  // tile visible to all waves
      FragBF a, bm0, bm1;
#pragma unroll
      for (int e = 0; e < 16; ++e) {
        int kk = e + ((e & 8) ? 8 : 0) + half * 8;   // K offset in slice
        a.u[e]   = abase[(size_t)(cb + kk) * astr];  // branchless (zero page)
        bm0.u[e] = ldsB[kk * 32 + mrow];
        bm1.u[e] = ldsB[kk * 32 + 16 + mrow];
      }
      acc0 = __builtin_amdgcn_wmma_f32_16x16x32_bf16(false, a.v, false, bm0.v,
                                                     (short)0, acc0, false, false);
      acc1 = __builtin_amdgcn_wmma_f32_16x16x32_bf16(false, a.v, false, bm1.v,
                                                     (short)0, acc1, false, false);
    }
  }
  float* ob0 = out + ((size_t)b * Cout + cch0) * N;
  float* ob1 = out + ((size_t)b * Cout + cch1) * N;
#pragma unroll
  for (int rr = 0; rr < 8; ++rr) {
    int p = tm * 16 + rr + 8 * half;
    if (p < N) {
      float y0 = acc0[rr] * scale0 + bias0;
      float y1 = acc1[rr] * scale1 + bias1;
      ob0[p] = LEAKY(y0);
      ob1[p] = LEAKY(y1);
    }
  }
}

// ---------------------------------------------------------------------------
// Generic bf16 GEMM: C[M][ldc] = A[M][K] @ B[K][Nc] (+ optional col scale).
// Block = 8 waves along M sharing one 16-col tile; B staged via async->LDS.
// A fragments load as two b128s on the fully-in-bounds fast path.
// ---------------------------------------------------------------------------
__global__ __launch_bounds__(256) void k_gemm_bf16(
    float* __restrict__ Cm, const unsigned short* __restrict__ A,
    const unsigned short* __restrict__ Bm, const float* __restrict__ colscale,
    const unsigned short* __restrict__ zpage,
    int M, int K, int Nc, int ldc) {
  const int TMt = (M + 15) >> 4;
  const int TNt = (Nc + 15) >> 4;
  __shared__ unsigned short ldsB[32 * 16];

  int lane = threadIdx.x & 31;
  int wloc = threadIdx.x >> 5;
  long blk = blockIdx.x;
  int tmb = (int)(blk / TNt);
  int tn  = (int)(blk % TNt);
  int tm  = tmb * 8 + wloc;

  int mrow = lane & 15, half = lane >> 4;
  int arow = tm * 16 + mrow;
  bool aval = (tm < TMt) && (arow < M);
  int col0 = tn * 16;
  int col  = col0 + mrow;
  bool cval = col < Nc;
  const unsigned short* abase = aval ? (A + (size_t)arow * K) : zpage;
  size_t astr = aval ? (size_t)1 : (size_t)0;

  int st_t  = threadIdx.x;
  int st_r  = st_t >> 1;
  int st_sg = st_t & 1;

  v8f acc = {0.f, 0.f, 0.f, 0.f, 0.f, 0.f, 0.f, 0.f};
  for (int k0 = 0; k0 < K; k0 += 32) {
    if (aval && k0 + 64 < K) __builtin_prefetch(abase + k0 + 64, 0, 0);
    __syncthreads();
    if (st_t < 64) {
      int kr = k0 + st_r;
      int cc = col0 + st_sg * 8;        // Nc is always a multiple of 8 here
      if (kr < K && cc + 7 < Nc) {
        async_copy_b128(&ldsB[st_r * 16 + st_sg * 8], &Bm[(size_t)kr * Nc + cc]);
      } else {
        uint4 z = {0u, 0u, 0u, 0u};
        *(uint4*)&ldsB[st_r * 16 + st_sg * 8] = z;
      }
    }
    async_wait_all();
    __syncthreads();
    FragBF a, bm;
    if (aval && (k0 + 32 <= K)) {
      // fast path: lane's 16 A elements = two contiguous 16B runs
      // (rows are 16B-aligned for K in {512, 1024, 5000})
      const uint4* ap = (const uint4*)(abase + k0 + half * 8);
      a.q[0] = ap[0];                   // kk = k0 + half*8 + [0..7]
      a.q[1] = ap[2];                   // kk = k0 + 16 + half*8 + [0..7]
    } else {
#pragma unroll
      for (int e = 0; e < 16; ++e) {
        int kk = k0 + e + ((e & 8) ? 8 : 0) + half * 8;
        int ki = kk < K ? kk : 0;
        unsigned short av = abase[(size_t)ki * astr];
        a.u[e] = (kk < K) ? av : (unsigned short)0;  // value mask, no branch
      }
    }
#pragma unroll
    for (int e = 0; e < 16; ++e) {
      int kk = e + ((e & 8) ? 8 : 0) + half * 8;
      bm.u[e] = ldsB[kk * 16 + mrow];
    }
    acc = __builtin_amdgcn_wmma_f32_16x16x32_bf16(false, a.v, false, bm.v,
                                                  (short)0, acc, false, false);
  }
  if (!cval) return;
  float cs = colscale ? colscale[col] : 1.f;
#pragma unroll
  for (int rr = 0; rr < 8; ++rr) {
    int row = tm * 16 + rr + 8 * half;
    if (row < M) Cm[(size_t)row * ldc + col] = acc[rr] * cs;
  }
}

// ---------------------------------------------------------------------------
// Clustering kernels
// ---------------------------------------------------------------------------
__global__ void k_row_argmax(int* ptr, const float* dec, int N) {
  int i = blockIdx.x;
  const float* row = dec + (size_t)i * N;
  __shared__ float sv[256];
  __shared__ int   si[256];
  float best = -3.0e38f; int bi = N;
  for (int n = threadIdx.x; n < N; n += 256) {
    float v = row[n];
    if (v > best || (v == best && n < bi)) { best = v; bi = n; }
  }
  sv[threadIdx.x] = best; si[threadIdx.x] = bi; __syncthreads();
  for (int s = 128; s > 0; s >>= 1) {
    if (threadIdx.x < s) {
      float ov = sv[threadIdx.x + s]; int oi = si[threadIdx.x + s];
      if (ov > sv[threadIdx.x] || (ov == sv[threadIdx.x] && oi < si[threadIdx.x])) {
        sv[threadIdx.x] = ov; si[threadIdx.x] = oi;
      }
    }
    __syncthreads();
  }
  if (threadIdx.x == 0) ptr[i] = (sv[0] > 0.8f) ? si[0] : i;
}

__global__ void k_ptr_jump(int* dst, const int* src, int N) {
  int i = blockIdx.x * blockDim.x + threadIdx.x;
  if (i < N) dst[i] = src[src[i]];
}
__global__ void k_flag_roots(int* flags, const int* ptr, int N) {
  int i = blockIdx.x * blockDim.x + threadIdx.x;
  if (i < N) flags[ptr[i]] = 1;
}
__global__ void k_scan_roots(int* rank, int* dC, const int* flags, int N) {
  if (threadIdx.x || blockIdx.x) return;
  int c = 0;
  for (int n = 0; n < N; ++n)
    if (flags[n]) { rank[n] = (c < CMAX) ? c : (CMAX - 1); ++c; }
  *dC = c < CMAX ? c : CMAX;
}
__global__ void k_labels(int* labels, const int* rank, const int* ptr, int N) {
  int i = blockIdx.x * blockDim.x + threadIdx.x;
  if (i < N) labels[i] = rank[ptr[i]];
}
__global__ void k_vcount(float* vc, const float* att, const int* labels, int N) {
  int i = blockIdx.x;
  int lab = labels[i]; if (lab > CMAX - 1) lab = CMAX - 1;
  const float* row = att + (size_t)i * N;
  float* dst = vc + (size_t)lab * N;
  for (int n = threadIdx.x; n < N; n += blockDim.x) atomicAdd(&dst[n], row[n]);
}
__global__ void k_cnt(float* cnt, const float* vc, const int* dC, int N) {
  int c = blockIdx.x; if (c >= *dC) return;
  __shared__ float red[256];
  const float* row = vc + (size_t)c * N;
  float s = 0.f;
  for (int n = threadIdx.x; n < N; n += 256) s += (row[n] > 0.f) ? 1.f : 0.f;
  red[threadIdx.x] = s; __syncthreads();
  for (int st = 128; st > 0; st >>= 1) {
    if (threadIdx.x < st) red[threadIdx.x] += red[threadIdx.x + st];
    __syncthreads();
  }
  if (threadIdx.x == 0) cnt[c] = red[0];
}
__global__ void k_score(float* score, const float* vc, const float* cnt,
                        const int* dC, int N) {
  int n = blockIdx.x * blockDim.x + threadIdx.x;
  if (n >= N) return;
  int C = *dC;
  float s = 0.f;
  for (int c = 0; c < C; ++c) {
    float d = cnt[c];
    s += vc[(size_t)c * N + n] / (d > 0.f ? d : 1.f);
  }
  score[n] = s;
}
__global__ void k_topk8(int* topk_idx, const float* vc, const int* dC, int N) {
  int c = blockIdx.x; if (c >= *dC) return;
  const float* row = vc + (size_t)c * N;
  __shared__ float sv[256];
  __shared__ int   si[256];
  __shared__ int   chosen[KTOP];
  for (int j = 0; j < KTOP; ++j) {
    float best = -3.0e38f; int bi = N;
    for (int n = threadIdx.x; n < N; n += 256) {
      bool skip = false;
      for (int t = 0; t < j; ++t) if (chosen[t] == n) skip = true;
      if (!skip) {
        float v = row[n];
        if (v > best || (v == best && n < bi)) { best = v; bi = n; }
      }
    }
    sv[threadIdx.x] = best; si[threadIdx.x] = bi; __syncthreads();
    for (int s = 128; s > 0; s >>= 1) {
      if (threadIdx.x < s) {
        float ov = sv[threadIdx.x + s]; int oi = si[threadIdx.x + s];
        if (ov > sv[threadIdx.x] || (ov == sv[threadIdx.x] && oi < si[threadIdx.x])) {
          sv[threadIdx.x] = ov; si[threadIdx.x] = oi;
        }
      }
      __syncthreads();
    }
    if (threadIdx.x == 0) { chosen[j] = si[0]; topk_idx[c * KTOP + j] = si[0]; }
    __syncthreads();
  }
}
__global__ void k_mark_vflags(int* vflags, const int* topk_idx, const int* dC) {
  int e = blockIdx.x * blockDim.x + threadIdx.x;
  if (e >= (*dC) * KTOP) return;
  vflags[topk_idx[e]] = 1;
}
__global__ void k_build_vtopk(int* vtopk, float* sv, int* dM, const int* vflags,
                              const float* score, int N) {
  if (threadIdx.x || blockIdx.x) return;
  int m = 0;
  for (int n = 0; n < N; ++n)
    if (vflags[n] && m < MMAX) { vtopk[m] = n; sv[m] = score[n]; ++m; }
  for (int j = m; j < MMAX; ++j) { vtopk[j] = 0; sv[j] = 0.f; }
  *dM = m;
}
__global__ void k_gather_rep(float* rep, const float* fvb, const int* vtopk,
                             const int* dM, int N) {
  int idx = blockIdx.x * blockDim.x + threadIdx.x;
  if (idx >= MMAX * 512) return;
  int mi = idx >> 9, ch = idx & 511;
  float v = 0.f;
  if (mi < *dM) v = fvb[(size_t)ch * N + vtopk[mi]];
  rep[idx] = v;
}

// ---------------------------------------------------------------------------
// Row softmax (in place); valid length optionally from device
// ---------------------------------------------------------------------------
__global__ void k_softmax_rows(float* X, int ld, int fixedlen, const int* dlen) {
  int row = blockIdx.x;
  int len = dlen ? *dlen : fixedlen;
  if (len <= 0) return;
  float* x = X + (size_t)row * ld;
  __shared__ float red[256];
  float mx = -3.0e38f;
  for (int j = threadIdx.x; j < len; j += 256) mx = fmaxf(mx, x[j]);
  red[threadIdx.x] = mx; __syncthreads();
  for (int s = 128; s > 0; s >>= 1) {
    if (threadIdx.x < s) red[threadIdx.x] = fmaxf(red[threadIdx.x], red[threadIdx.x + s]);
    __syncthreads();
  }
  mx = red[0]; __syncthreads();
  float sum = 0.f;
  for (int j = threadIdx.x; j < len; j += 256) { float e = expf(x[j] - mx); x[j] = e; sum += e; }
  red[threadIdx.x] = sum; __syncthreads();
  for (int s = 128; s > 0; s >>= 1) {
    if (threadIdx.x < s) red[threadIdx.x] += red[threadIdx.x + s];
    __syncthreads();
  }
  float inv = 1.f / red[0];
  for (int j = threadIdx.x; j < len; j += 256) x[j] *= inv;
}

// ---------------------------------------------------------------------------
// seq_conv: 3-tap 1D conv over M rows + BN + leaky (512 ch)
// ---------------------------------------------------------------------------
__global__ void k_seq_conv_bn(float* __restrict__ out, const float* __restrict__ xin,
                              const float* __restrict__ wps,
                              const float* g, const float* bb,
                              const float* mm, const float* vv, const int* dM) {
  int m = blockIdx.x;
  int Mv = *dM;
  if (m >= Mv) return;
  __shared__ float xs[3 * 512];
  for (int i = threadIdx.x; i < 3 * 512; i += blockDim.x) {
    int dt = i / 512, ci = i % 512;
    int mr = m + dt - 1;
    xs[i] = (mr >= 0 && mr < Mv) ? xin[(size_t)mr * 512 + ci] : 0.f;
  }
  __syncthreads();
  for (int co = threadIdx.x; co < 512; co += blockDim.x) {
    float acc = 0.f;
    for (int k = 0; k < 3 * 512; ++k) acc = fmaf(xs[k], wps[(size_t)k * 512 + co], acc);
    float sc = g[co] * rsqrtf(vv[co] + BN_EPS);
    float bi = bb[co] - mm[co] * sc;
    float y = acc * sc + bi;
    out[(size_t)m * 512 + co] = LEAKY(y);
  }
}

// ---------------------------------------------------------------------------
// Misc pipeline glue
// ---------------------------------------------------------------------------
__global__ void k_dist(float* dist, const float* fqb, const float* ebig, int N) {
  long idx = (long)blockIdx.x * blockDim.x + threadIdx.x;
  if (idx >= (long)512 * N) return;
  int ch = (int)(idx / N), n = (int)(idx % N);
  dist[idx] = fqb[idx] + ebig[(size_t)n * 512 + ch];
}
__global__ void k_f_to_fbig(unsigned short* fbig, const float* f, int N) {
  long idx = (long)blockIdx.x * blockDim.x + threadIdx.x;
  if (idx >= (long)NB * 512 * N) return;
  int b = (int)(idx / ((long)512 * N));
  long rem = idx - (long)b * 512 * N;
  int ch = (int)(rem / N), n = (int)(rem % N);
  fbig[((size_t)b * 1024 + ch) * N + n] = f2bf_u(f[idx]);
}
__global__ void k_ctx_to_fbig(unsigned short* fbig, const float* ctx, int b, int N) {
  long idx = (long)blockIdx.x * blockDim.x + threadIdx.x;
  if (idx >= (long)512 * N) return;
  int ch = (int)(idx / N), n = (int)(idx % N);
  fbig[((size_t)b * 1024 + 512 + ch) * N + n] = f2bf_u(ctx[idx]);
}

// ---------------------------------------------------------------------------
// Host launcher
// ---------------------------------------------------------------------------
static inline int blks(long n, int t) { return (int)((n + t - 1) / t); }

extern "C" void kernel_launch(void* const* d_in, const int* in_sizes, int n_in,
                              void* d_out, int out_size, void* d_ws, size_t ws_size,
                              hipStream_t stream) {
  (void)in_sizes; (void)n_in; (void)out_size; (void)ws_size;
  const int N = NPix, H = HH, W = WW;

  const float* feats = (const float*)d_in[0];
  const float* attn  = (const float*)d_in[1];
  const float* dec   = (const float*)d_in[2];
  const int IP_FEATS = 3, IP_QUERY = 8, IP_VALUE = 13, IP_COLLECT = 18,
            IP_INTERACT = 23, IP_DISTRIB = 28, IP_FINAL = 33;
  auto P = [&](int base, int off) { return (const float*)d_in[base + off]; };

  // ---- workspace arena ----
  char* ws = (char*)d_ws;
  size_t off = 0;
  auto alloc = [&](size_t bytes) -> void* {
    void* p = ws + off; off += (bytes + 255) & ~(size_t)255; return p;
  };
  unsigned short* zpage    = (unsigned short*)alloc(256);
  unsigned short* feats_bf = (unsigned short*)alloc((size_t)NB * 256 * N * 2);
  unsigned short* wp_feats = (unsigned short*)alloc((size_t)9 * 256 * 512 * 2);
  unsigned short* wp_query = (unsigned short*)alloc((size_t)9 * 512 * 512 * 2);
  unsigned short* wp_value = (unsigned short*)alloc((size_t)9 * 512 * 512 * 2);
  unsigned short* wp_dist  = (unsigned short*)alloc((size_t)9 * 512 * 512 * 2);
  unsigned short* wp_final = (unsigned short*)alloc((size_t)9 * 1024 * 512 * 2);
  float* wseq_col = (float*)alloc((size_t)3 * 512 * 512 * 4);
  float* wseq_int = (float*)alloc((size_t)3 * 512 * 512 * 4);
  float* f_f32  = (float*)alloc((size_t)NB * 512 * N * 4);
  unsigned short* f_bf  = (unsigned short*)alloc((size_t)NB * 512 * N * 2);
  float* fq_f32 = (float*)alloc((size_t)NB * 512 * N * 4);
  float* fv_f32 = (float*)alloc((size_t)NB * 512 * N * 4);
  unsigned short* fv_bf = (unsigned short*)alloc((size_t)NB * 512 * N * 2);
  unsigned short* fbig_bf = (unsigned short*)alloc((size_t)NB * 1024 * N * 2);
  int* ptrA  = (int*)alloc((size_t)N * 4);
  int* ptrB  = (int*)alloc((size_t)N * 4);
  int* flags = (int*)alloc((size_t)N * 4);
  int* rankb = (int*)alloc((size_t)N * 4);
  int* labels = (int*)alloc((size_t)N * 4);
  int* dC = (int*)alloc(4);
  int* dM = (int*)alloc(4);
  float* vcount = (float*)alloc((size_t)CMAX * N * 4);
  float* cnt    = (float*)alloc((size_t)CMAX * 4);
  float* score  = (float*)alloc((size_t)N * 4);
  int* topk_idx = (int*)alloc((size_t)CMAX * KTOP * 4);
  int* vflags   = (int*)alloc((size_t)N * 4);
  int* vtopk    = (int*)alloc((size_t)MMAX * 4);
  float* svv    = (float*)alloc((size_t)MMAX * 4);
  float* rep    = (float*)alloc((size_t)MMAX * 512 * 4);
  unsigned short* rep_bf = (unsigned short*)alloc((size_t)MMAX * 512 * 2);
  unsigned short* fvT_bf = (unsigned short*)alloc((size_t)N * 512 * 2);
  unsigned short* fqT_bf = (unsigned short*)alloc((size_t)N * 512 * 2);
  float* cw = (float*)alloc((size_t)MMAX * N * 4);
  unsigned short* cw_bf = (unsigned short*)alloc((size_t)MMAX * N * 2);
  float* e_small = (float*)alloc((size_t)MMAX * 512 * 4);
  float* cfeat   = (float*)alloc((size_t)MMAX * 512 * 4);
  float* crep    = (float*)alloc((size_t)MMAX * 512 * 4);
  unsigned short* crep_bf  = (unsigned short*)alloc((size_t)MMAX * 512 * 2);
  unsigned short* crepT_bf = (unsigned short*)alloc((size_t)512 * MMAX * 2);
  float* iw = (float*)alloc((size_t)MMAX * MMAX * 4);
  unsigned short* iw_bf = (unsigned short*)alloc((size_t)MMAX * MMAX * 2);
  float* ifeat = (float*)alloc((size_t)MMAX * 512 * 4);
  float* irep  = (float*)alloc((size_t)MMAX * 512 * 4);
  unsigned short* irep_bf  = (unsigned short*)alloc((size_t)MMAX * 512 * 2);
  unsigned short* irepT_bf = (unsigned short*)alloc((size_t)512 * MMAX * 2);
  float* dw = (float*)alloc((size_t)N * MMAX * 4);
  unsigned short* dw_bf = (unsigned short*)alloc((size_t)N * MMAX * 2);
  float* e_big = (float*)alloc((size_t)N * 512 * 4);
  float* dist  = (float*)alloc((size_t)512 * N * 4);
  unsigned short* dist_bf = (unsigned short*)alloc((size_t)512 * N * 2);
  float* ctx = (float*)alloc((size_t)512 * N * 4);

  // ---- launch helpers ----
  auto conv = [&](float* o, const unsigned short* x, const unsigned short* wp,
                  int pb, int nb, int Cin, int Cout) {
    int TM = (N + 15) / 16, TMB = (TM + 7) / 8, TN = Cout / 32;
    long blocks = (long)nb * TMB * TN;
    k_conv_bn<<<(int)blocks, 256, 0, stream>>>(
        o, x, wp, P(pb, 1), P(pb, 2), P(pb, 3), P(pb, 4), zpage,
        nb, Cin, Cout, H, W);
  };
  auto gemm = [&](float* C, const unsigned short* A, const unsigned short* B,
                  const float* cs, int M, int K, int Nc, int ldc) {
    int TMt = (M + 15) / 16, TMB = (TMt + 7) / 8, TNt = (Nc + 15) / 16;
    long blocks = (long)TMB * TNt;
    k_gemm_bf16<<<(int)blocks, 256, 0, stream>>>(C, A, B, cs, zpage, M, K, Nc, ldc);
  };
  auto cvt = [&](unsigned short* d, const float* s, long n) {
    k_f32_to_bf16<<<blks(n, 256), 256, 0, stream>>>(d, s, n);
  };
  auto fillf = [&](float* p, float v, long n) {
    k_fill_f32<<<blks(n, 256), 256, 0, stream>>>(p, v, n);
  };
  auto filli = [&](int* p, int v, long n) {
    k_fill_i32<<<blks(n, 256), 256, 0, stream>>>(p, v, n);
  };

  // ---- stage 0: pack weights, convert input, zero page ----
  filli((int*)zpage, 0, 64);
  cvt(feats_bf, feats, (long)NB * 256 * N);
  k_pack_conv_w<<<blks((long)512 * 256 * 9, 256), 256, 0, stream>>>(wp_feats, P(IP_FEATS, 0), 512, 256);
  k_pack_conv_w<<<blks((long)512 * 512 * 9, 256), 256, 0, stream>>>(wp_query, P(IP_QUERY, 0), 512, 512);
  k_pack_conv_w<<<blks((long)512 * 512 * 9, 256), 256, 0, stream>>>(wp_value, P(IP_VALUE, 0), 512, 512);
  k_pack_conv_w<<<blks((long)512 * 512 * 9, 256), 256, 0, stream>>>(wp_dist, P(IP_DISTRIB, 0), 512, 512);
  k_pack_conv_w<<<blks((long)512 * 1024 * 9, 256), 256, 0, stream>>>(wp_final, P(IP_FINAL, 0), 512, 1024);
  k_pack_seq_w<<<blks((long)512 * 512 * 3, 256), 256, 0, stream>>>(wseq_col, P(IP_COLLECT, 0));
  k_pack_seq_w<<<blks((long)512 * 512 * 3, 256), 256, 0, stream>>>(wseq_int, P(IP_INTERACT, 0));

  // ---- stage A: base convs ----
  conv(f_f32, feats_bf, wp_feats, IP_FEATS, NB, 256, 512);
  cvt(f_bf, f_f32, (long)NB * 512 * N);
  conv(fq_f32, f_bf, wp_query, IP_QUERY, NB, 512, 512);
  conv(fv_f32, f_bf, wp_value, IP_VALUE, NB, 512, 512);
  cvt(fv_bf, fv_f32, (long)NB * 512 * N);
  k_f_to_fbig<<<blks((long)NB * 512 * N, 256), 256, 0, stream>>>(fbig_bf, f_f32, N);

  // ---- stage C: per-batch attention/context ----
  for (int b = 0; b < NB; ++b) {
    const float* decb = dec + (size_t)b * N * N;
    const float* attb = attn + (size_t)b * N * N;
    const float* fvb  = fv_f32 + (size_t)b * 512 * N;
    const float* fqb  = fq_f32 + (size_t)b * 512 * N;
    const unsigned short* fvb_bf = fv_bf + (size_t)b * 512 * N;

    // clustering
    k_row_argmax<<<N, 256, 0, stream>>>(ptrA, decb, N);
    for (int it = 0; it < 10; ++it) {
      int* s = (it & 1) ? ptrB : ptrA;
      int* d = (it & 1) ? ptrA : ptrB;
      k_ptr_jump<<<blks(N, 256), 256, 0, stream>>>(d, s, N);
    }
    filli(flags, 0, N);
    k_flag_roots<<<blks(N, 256), 256, 0, stream>>>(flags, ptrA, N);
    k_scan_roots<<<1, 1, 0, stream>>>(rankb, dC, flags, N);
    k_labels<<<blks(N, 256), 256, 0, stream>>>(labels, rankb, ptrA, N);

    // segment sum + score + representatives
    fillf(vcount, 0.f, (long)CMAX * N);
    k_vcount<<<N, 256, 0, stream>>>(vcount, attb, labels, N);
    k_cnt<<<CMAX, 256, 0, stream>>>(cnt, vcount, dC, N);
    k_score<<<blks(N, 256), 256, 0, stream>>>(score, vcount, cnt, dC, N);
    k_topk8<<<CMAX, 256, 0, stream>>>(topk_idx, vcount, dC, N);
    filli(vflags, 0, N);
    k_mark_vflags<<<blks((long)CMAX * KTOP, 256), 256, 0, stream>>>(vflags, topk_idx, dC);
    k_build_vtopk<<<1, 1, 0, stream>>>(vtopk, svv, dM, vflags, score, N);
    k_gather_rep<<<blks((long)MMAX * 512, 256), 256, 0, stream>>>(rep, fvb, vtopk, dM, N);
    cvt(rep_bf, rep, (long)MMAX * 512);
    k_transpose_bf<<<blks((long)512 * N, 256), 256, 0, stream>>>(fvT_bf, fvb, 512, N);
    k_transpose_bf<<<blks((long)512 * N, 256), 256, 0, stream>>>(fqT_bf, fqb, 512, N);

    // collect: cw = softmax(rep @ fv * score); cfeat = rep + cw @ fv.T
    gemm(cw, rep_bf, fvb_bf, score, MMAX, 512, N, N);
    k_softmax_rows<<<MMAX, 256, 0, stream>>>(cw, N, N, (const int*)nullptr);
    cvt(cw_bf, cw, (long)MMAX * N);
    gemm(e_small, cw_bf, fvT_bf, nullptr, MMAX, N, 512, 512);
    k_add_f32<<<blks((long)MMAX * 512, 256), 256, 0, stream>>>(cfeat, rep, e_small, (long)MMAX * 512);
    fillf(crep, 0.f, (long)MMAX * 512);
    k_seq_conv_bn<<<MMAX, 256, 0, stream>>>(crep, cfeat, wseq_col,
        P(IP_COLLECT, 1), P(IP_COLLECT, 2), P(IP_COLLECT, 3), P(IP_COLLECT, 4), dM);
    cvt(crep_bf, crep, (long)MMAX * 512);
    k_transpose_bf<<<blks((long)MMAX * 512, 256), 256, 0, stream>>>(crepT_bf, crep, MMAX, 512);

    // interact: iw = softmax(crep @ (crep.T * sv)); irep = seq(crep + iw@crep)
    gemm(iw, crep_bf, crepT_bf, svv, MMAX, 512, MMAX, MMAX);
    k_softmax_rows<<<MMAX, 256, 0, stream>>>(iw, MMAX, MMAX, dM);
    cvt(iw_bf, iw, (long)MMAX * MMAX);
    gemm(e_small, iw_bf, crep_bf, nullptr, MMAX, MMAX, 512, 512);
    k_add_f32<<<blks((long)MMAX * 512, 256), 256, 0, stream>>>(ifeat, crep, e_small, (long)MMAX * 512);
    fillf(irep, 0.f, (long)MMAX * 512);
    k_seq_conv_bn<<<MMAX, 256, 0, stream>>>(irep, ifeat, wseq_int,
        P(IP_INTERACT, 1), P(IP_INTERACT, 2), P(IP_INTERACT, 3), P(IP_INTERACT, 4), dM);
    cvt(irep_bf, irep, (long)MMAX * 512);
    k_transpose_bf<<<blks((long)MMAX * 512, 256), 256, 0, stream>>>(irepT_bf, irep, MMAX, 512);

    // distribute: dw = softmax(fq.T @ (irep.T * sv)); dist = fq + (dw@irep).T
    gemm(dw, fqT_bf, irepT_bf, svv, N, 512, MMAX, MMAX);
    k_softmax_rows<<<N, 256, 0, stream>>>(dw, MMAX, MMAX, dM);
    cvt(dw_bf, dw, (long)N * MMAX);
    gemm(e_big, dw_bf, irep_bf, nullptr, N, MMAX, 512, 512);
    k_dist<<<blks((long)512 * N, 256), 256, 0, stream>>>(dist, fqb, e_big, N);
    cvt(dist_bf, dist, (long)512 * N);
    conv(ctx, dist_bf, wp_dist, IP_DISTRIB, 1, 512, 512);
    k_ctx_to_fbig<<<blks((long)512 * N, 256), 256, 0, stream>>>(fbig_bf, ctx, b, N);
  }

  // ---- stage D: final conv over concat[f, ctx] ----
  conv((float*)d_out, fbig_bf, wp_final, IP_FINAL, NB, 1024, 512);
}
